// ARModel_29944511988218
// MI455X (gfx1250) — matrix operand order repaired and checked by reading, not compiled, not run on hardware
//
#include <hip/hip_runtime.h>

// ---------------------------------------------------------------------------
// GRU seq2seq on gfx1250 (MI455X), persistent-kernel design.
//   B=128, I=256, H=1024, S=1024 encode steps, T=256 decode steps.
//   Per step: gates[128,3072] = x·W_ih^T + h·W_hh^T (+biases), GRU update.
//   bf16 WMMA (v_wmma_f32_16x16x32_bf16), f32 accumulate, h kept in f32 regs.
//   Weights pre-swizzled to WMMA B-fragment layout and cached in LDS (152KB/WG,
//   reused 1280x). Grid-wide barrier via atomic counter (64 resident WGs).
//   If ws_size permits, x is pre-converted to bf16 once (removes per-step cvt).
// ---------------------------------------------------------------------------

typedef __bf16 bf16_t;
typedef __attribute__((ext_vector_type(16))) __bf16 v16bf;
typedef __attribute__((ext_vector_type(8)))  float  v8f;

#define NWG        64
#define LDS_WIH    0          //  8 kt * 3 gates * 32 lanes * 32B = 24576 B
#define LDS_WHH    24576      // 32 kt * 3 gates * 32 lanes * 32B = 98304 B
#define LDS_WLIN   122880     // 32 kt * 32 lanes * 32B           = 32768 B
#define LDS_TOTAL  155648

#define WS_BASE    9000960UL  // weights/h/y/counter region (rounded up)
#define XBF_BYTES  67108864UL // 128*1024*256 bf16

__device__ __forceinline__ v8f wmma_bf16(v16bf a, v16bf b, v8f c) {
    return __builtin_amdgcn_wmma_f32_16x16x32_bf16(
        /*neg_a=*/false, a, /*neg_b=*/false, b,
        /*c_mod=*/(short)0, c, /*reuse_a=*/false, /*reuse_b=*/false);
}

__device__ __forceinline__ float fast_sigmoid(float x) {
    return 1.0f / (1.0f + __expf(-x));
}
__device__ __forceinline__ float fast_tanh(float x) {
    return 1.0f - 2.0f / (__expf(2.0f * x) + 1.0f);
}

// A fragment (16x32 MxK, bf16 row-major src, row stride rs elements).
// lanes 0-15: M=lane, K 0..7 in v0-3, 16..23 in v4-7
// lanes16-31: M=lane-16, K 8..15 / 24..31   -> col offset = halfk*8 (+16)
__device__ __forceinline__ v16bf load_a_bf16(const bf16_t* base, long rs,
                                             int row, int halfk, int kt) {
    const bf16_t* p = base + (long)row * rs + kt * 32 + halfk * 8;
    union { v16bf v; uint4 q[2]; } u;
    u.q[0] = *(const uint4*)(p);
    u.q[1] = *(const uint4*)(p + 16);
    return u.v;
}

// Same fragment, but from an f32 row-major source (fallback path), cvt->bf16.
__device__ __forceinline__ v16bf load_a_f32(const float* base, long rs,
                                            int row, int halfk, int kt) {
    const float* p = base + (long)row * rs + kt * 32 + halfk * 8;
    float4 a0 = *(const float4*)(p);
    float4 a1 = *(const float4*)(p + 4);
    float4 b0 = *(const float4*)(p + 16);
    float4 b1 = *(const float4*)(p + 20);
    float t[16] = {a0.x, a0.y, a0.z, a0.w, a1.x, a1.y, a1.z, a1.w,
                   b0.x, b0.y, b0.z, b0.w, b1.x, b1.y, b1.z, b1.w};
    union { v16bf v; bf16_t e[16]; } u;
#pragma unroll
    for (int j = 0; j < 16; j++) u.e[j] = (bf16_t)t[j];
    return u.v;
}

// B fragment from LDS: 32 bytes per lane, contiguous, pre-swizzled.
__device__ __forceinline__ v16bf load_b_lds(const char* smem, int byteoff) {
    union { v16bf v; uint4 q[2]; } u;
    u.q[0] = *(const uint4*)(smem + byteoff);
    u.q[1] = *(const uint4*)(smem + byteoff + 16);
    return u.v;
}

// Device-wide barrier: monotonically increasing counter, 1 add per WG.
__device__ __forceinline__ void grid_sync(unsigned* cnt, unsigned target) {
    __threadfence();              // release: every wave flushes its stores
    __syncthreads();
    if (threadIdx.x == 0) {
        atomicAdd(cnt, 1u);
        while (__hip_atomic_load(cnt, __ATOMIC_RELAXED,
                                 __HIP_MEMORY_SCOPE_AGENT) < target) {
            __builtin_amdgcn_s_sleep(2);
        }
    }
    __syncthreads();
    __threadfence();              // acquire: invalidate stale near-cache lines
}

// One GRU step for this wave's (mtile, ntile) 16x16 tile of h.
// XF32: x source is f32 (fallback) vs bf16 (xbf / decode ybf).
template <bool XF32>
__device__ __forceinline__ void gru_step(
    const void* xbase, long xrs,          // x source, row stride (elements)
    const bf16_t* hsrc, bf16_t* hdst,     // hbf[rp] read, hbf[rp^1] write
    const char* smem,
    const float* b_ih, const float* b_hh,
    int mtile, int ntile, int lane, float hold[8]) {

    const int c     = (lane & 15);
    const int col   = ntile * 16 + c;     // column in [0, H)
    const int halfk = lane >> 4;
    const int row0  = mtile * 16 + c;     // A-fragment source row

    const float br  = b_ih[col]        + b_hh[col];
    const float bz  = b_ih[1024 + col] + b_hh[1024 + col];
    const float bin = b_ih[2048 + col];
    const float bhn = b_hh[2048 + col];

    v8f accr, accz, accin, acchn;
#pragma unroll
    for (int j = 0; j < 8; j++) {
        accr[j] = br; accz[j] = bz; accin[j] = bin; acchn[j] = bhn;
    }

    // ---- x contribution: K = 256 (8 ktiles), B-frags from LDS W_ih region --
#pragma unroll
    for (int kt = 0; kt < 8; kt++) {
        v16bf a = XF32 ? load_a_f32((const float*)xbase, xrs, row0, halfk, kt)
                       : load_a_bf16((const bf16_t*)xbase, xrs, row0, halfk, kt);
        int base = LDS_WIH + ((kt * 3) * 32 + lane) * 32;
        accr  = wmma_bf16(a, load_b_lds(smem, base),            accr);
        accz  = wmma_bf16(a, load_b_lds(smem, base + 32 * 32),  accz);
        accin = wmma_bf16(a, load_b_lds(smem, base + 64 * 32),  accin);
    }

    // ---- h contribution: K = 1024 (32 ktiles), B-frags from LDS W_hh -------
#pragma unroll 4
    for (int kt = 0; kt < 32; kt++) {
        v16bf a = load_a_bf16(hsrc, 1024, row0, halfk, kt);
        int base = LDS_WHH + ((kt * 3) * 32 + lane) * 32;
        accr  = wmma_bf16(a, load_b_lds(smem, base),            accr);
        accz  = wmma_bf16(a, load_b_lds(smem, base + 32 * 32),  accz);
        acchn = wmma_bf16(a, load_b_lds(smem, base + 64 * 32),  acchn);
    }

    // ---- gate math + update; h_old lives in registers (same owner wave) ----
#pragma unroll
    for (int j = 0; j < 8; j++) {
        float r = fast_sigmoid(accr[j]);
        float z = fast_sigmoid(accz[j]);
        float n = fast_tanh(accin[j] + r * acchn[j]);
        float h = (1.0f - z) * n + z * hold[j];
        hold[j] = h;
        int row = mtile * 16 + j + halfk * 8;   // C-layout row of v_j
        hdst[(long)row * 1024 + col] = (bf16_t)h;
    }
}

// y = h @ W_lin^T + b_lin for tile (mtile, nt). Writes bf16 ybf, opt f32 out.
__device__ __forceinline__ void project(
    const bf16_t* hsrc, const char* smem, const float* b_lin,
    float* outp, bf16_t* ybf, int mtile, int nt, int lane, int t) {

    const int c     = (lane & 15);
    const int col   = nt * 16 + c;
    const int halfk = lane >> 4;
    const int row0  = mtile * 16 + c;

    const float bv = b_lin[col];
    v8f acc;
#pragma unroll
    for (int j = 0; j < 8; j++) acc[j] = bv;

#pragma unroll 4
    for (int kt = 0; kt < 32; kt++) {
        v16bf a = load_a_bf16(hsrc, 1024, row0, halfk, kt);
        v16bf b = load_b_lds(smem, LDS_WLIN + (kt * 32 + lane) * 32);
        acc = wmma_bf16(a, b, acc);
    }
#pragma unroll
    for (int j = 0; j < 8; j++) {
        int row = mtile * 16 + j + halfk * 8;
        float y = acc[j];
        ybf[(long)row * 256 + col] = (bf16_t)y;
        if (t >= 0)  // out[b, t, i], B=128 T=256 I=256
            outp[(long)row * 65536 + (long)t * 256 + col] = y;
    }
}

// ---------------------------------------------------------------------------
// Prep kernels
// ---------------------------------------------------------------------------

// Zero h0 (bf16 buffer 0) and the barrier counter.
__global__ void k_zero(bf16_t* hbf, unsigned* cnt) {
    long i = (long)blockIdx.x * 256 + threadIdx.x;
    if (i < 131072) hbf[i] = (bf16_t)0.0f;
    if (i == 0) *cnt = 0u;
}

// Elementwise f32 -> bf16 (x pre-conversion), 4 elements per thread.
__global__ void k_cvt(const float* __restrict__ src, bf16_t* __restrict__ dst,
                      long total) {
    long i = ((long)blockIdx.x * 256 + threadIdx.x) * 4;
    if (i >= total) return;
    float4 v = *(const float4*)(src + i);
    dst[i + 0] = (bf16_t)v.x;
    dst[i + 1] = (bf16_t)v.y;
    dst[i + 2] = (bf16_t)v.z;
    dst[i + 3] = (bf16_t)v.w;
}

// Swizzle f32 weight matrix W[Nout, Kin] (row-major) into bf16 WMMA
// B-fragments: dst[((kt*NT + nt)*32 + lane)*16 + j] where
//   k = kt*32 + (lane>=16 ? 8 : 0) + (j&7) + (j>=8 ? 16 : 0),  n = nt*16 + (lane&15).
__global__ void k_pack(const float* __restrict__ W, bf16_t* __restrict__ dst,
                       int Kin, int NT, long total) {
    long e = (long)blockIdx.x * 256 + threadIdx.x;
    if (e >= total) return;
    int  j    = (int)(e & 15);
    int  lane = (int)((e >> 4) & 31);
    long rest = e >> 9;
    int  nt   = (int)(rest % NT);
    int  kt   = (int)(rest / NT);
    int  k = kt * 32 + (lane >> 4) * 8 + (j & 7) + (j >> 3) * 16;
    int  n = nt * 16 + (lane & 15);
    dst[e] = (bf16_t)W[(long)n * Kin + k];
}

// ---------------------------------------------------------------------------
// Persistent main kernel: 64 WGs x 256 threads (8 waves).
//   ntile = blockIdx.x (slice of H), mtile = wave index (slice of batch).
// ---------------------------------------------------------------------------
__global__ void __launch_bounds__(256) gru_main(
    const float* __restrict__ x,
    const bf16_t* __restrict__ xbf,      // pre-converted x (may be null)
    int use_xbf,
    const float* __restrict__ b_ih, const float* __restrict__ b_hh,
    const float* __restrict__ b_lin,
    const bf16_t* __restrict__ wihf, const bf16_t* __restrict__ whhf,
    const bf16_t* __restrict__ wlinf,
    bf16_t* hbf, bf16_t* ybf, unsigned* cnt, float* out) {

    extern __shared__ char smem[];
    const int ntile = blockIdx.x;
    const int lane  = threadIdx.x & 31;
    const int mtile = threadIdx.x >> 5;

    // ---- cache this WG's weight fragment slices in LDS (reused 1280x) ------
    for (int i = threadIdx.x; i < 1536; i += 256) {     // W_ih: 24 KB
        int p = i & 1, ln = (i >> 1) & 31, fg = i >> 6;
        int g = fg % 3, kt = fg / 3;
        long src = (((long)kt * 192 + (ntile + g * 64)) * 32 + ln) * 16 + p * 8;
        ((uint4*)(smem + LDS_WIH))[i] = *(const uint4*)(wihf + src);
    }
    for (int i = threadIdx.x; i < 6144; i += 256) {     // W_hh: 96 KB
        int p = i & 1, ln = (i >> 1) & 31, fg = i >> 6;
        int g = fg % 3, kt = fg / 3;
        long src = (((long)kt * 192 + (ntile + g * 64)) * 32 + ln) * 16 + p * 8;
        ((uint4*)(smem + LDS_WHH))[i] = *(const uint4*)(whhf + src);
    }
    if (ntile < 16) {
        for (int i = threadIdx.x; i < 2048; i += 256) { // W_lin: 32 KB
            int p = i & 1, ln = (i >> 1) & 31, kt = i >> 6;
            long src = (((long)kt * 16 + ntile) * 32 + ln) * 16 + p * 8;
            ((uint4*)(smem + LDS_WLIN))[i] = *(const uint4*)(wlinf + src);
        }
    }
    __syncthreads();

    float hold[8];
#pragma unroll
    for (int j = 0; j < 8; j++) hold[j] = 0.0f;   // h0 = 0

    unsigned epoch = 0;
    int rp = 0;                                    // hbf read parity

    // ---- encode: 1024 steps over x -----------------------------------------
    for (int s = 0; s < 1024; s++) {
        if (use_xbf)
            gru_step<false>(xbf + (long)s * 256, 262144L,   // xbf[:, s, :]
                            hbf + rp * 131072, hbf + (rp ^ 1) * 131072,
                            smem, b_ih, b_hh, mtile, ntile, lane, hold);
        else
            gru_step<true>(x + (long)s * 256, 262144L,      // x[:, s, :]
                           hbf + rp * 131072, hbf + (rp ^ 1) * 131072,
                           smem, b_ih, b_hh, mtile, ntile, lane, hold);
        epoch++; grid_sync(cnt, epoch * (unsigned)NWG);
        rp ^= 1;
    }

    // ---- y0 = h_n @ W_lin^T + b_lin (feeds decode, not stored to out) ------
    if (ntile < 16)
        project(hbf + rp * 131072, smem, b_lin, out, ybf, mtile, ntile, lane, -1);
    epoch++; grid_sync(cnt, epoch * (unsigned)NWG);

    // ---- decode: 256 autoregressive steps ----------------------------------
    for (int t = 0; t < 256; t++) {
        gru_step<false>(ybf, 256L,
                        hbf + rp * 131072, hbf + (rp ^ 1) * 131072,
                        smem, b_ih, b_hh, mtile, ntile, lane, hold);
        epoch++; grid_sync(cnt, epoch * (unsigned)NWG);
        rp ^= 1;
        if (ntile < 16)
            project(hbf + rp * 131072, smem, b_lin, out, ybf, mtile, ntile, lane, t);
        epoch++; grid_sync(cnt, epoch * (unsigned)NWG);
    }
}

// ---------------------------------------------------------------------------
extern "C" void kernel_launch(void* const* d_in, const int* in_sizes, int n_in,
                              void* d_out, int out_size, void* d_ws, size_t ws_size,
                              hipStream_t stream) {
    (void)in_sizes; (void)n_in; (void)out_size;

    const float* x     = (const float*)d_in[0];
    // d_in[1] = target_seq_len (== 256, compile-time constant here)
    const float* W_ih  = (const float*)d_in[2];
    const float* W_hh  = (const float*)d_in[3];
    const float* b_ih  = (const float*)d_in[4];
    const float* b_hh  = (const float*)d_in[5];
    const float* W_lin = (const float*)d_in[6];
    const float* b_lin = (const float*)d_in[7];
    float* out = (float*)d_out;

    // workspace layout (~9 MB fixed + optional 64 MB x-bf16 staging)
    char* w = (char*)d_ws;
    size_t o = 0;
    bf16_t* whhf = (bf16_t*)(w + o); o += 6291456;   // 3072x1024 bf16 frags
    bf16_t* wihf = (bf16_t*)(w + o); o += 1572864;   // 3072x256
    bf16_t* wlinf= (bf16_t*)(w + o); o += 524288;    // 256x1024
    bf16_t* hbf  = (bf16_t*)(w + o); o += 524288;    // 2 x 128x1024 (dbl buf)
    bf16_t* ybf  = (bf16_t*)(w + o); o += 65536;     // 128x256
    unsigned* cnt= (unsigned*)(w + o); o += 256;

    const int use_xbf = (ws_size >= WS_BASE + XBF_BYTES) ? 1 : 0;
    bf16_t* xbf = (bf16_t*)(w + WS_BASE);

    k_zero<<<512, 256, 0, stream>>>(hbf, cnt);
    k_pack<<<(3145728 + 255) / 256, 256, 0, stream>>>(W_hh,  whhf, 1024, 192, 3145728L);
    k_pack<<<(786432  + 255) / 256, 256, 0, stream>>>(W_ih,  wihf,  256, 192,  786432L);
    k_pack<<<(262144  + 255) / 256, 256, 0, stream>>>(W_lin, wlinf, 1024,  16,  262144L);
    if (use_xbf) {
        // 128*1024*256 = 33554432 elements, 4 per thread
        k_cvt<<<32768, 256, 0, stream>>>(x, xbf, 33554432L);
    }

    (void)hipFuncSetAttribute((const void*)gru_main,
                              hipFuncAttributeMaxDynamicSharedMemorySize,
                              LDS_TOTAL);
    gru_main<<<NWG, 256, LDS_TOTAL, stream>>>(x, xbf, use_xbf,
                                              b_ih, b_hh, b_lin,
                                              wihf, whhf, wlinf,
                                              hbf, ybf, cnt, out);
}
